// FFTMixer_58523224375530
// MI455X (gfx1250) — compile-verified, matrix operand order checked by reading, not run on hardware
//
#include <hip/hip_runtime.h>
#include <hip/hip_bf16.h>
#include <math.h>

// ---------------------------------------------------------------------------
// FFT mixer for MI455X (gfx1250), wave32 + WMMA f16 16x16x32.
// 768-pt FFT factored 16x48; both DFT stages run as v_wmma_f32_16x16x32_f16
// GEMMs against constant DFT fragments held in LDS. Fully fused per row:
//   load row -> FFT -> adaptive filter -> modReLU -> iFFT -> store real part.
// Input rows are streamed with GLOBAL_LOAD_ASYNC_TO_LDS_B128, double-buffered
// across row iterations (ASYNCcnt pipeline).
// ---------------------------------------------------------------------------

typedef __attribute__((ext_vector_type(16))) _Float16 v16h;
typedef __attribute__((ext_vector_type(8)))  _Float16 v8h;
typedef __attribute__((ext_vector_type(8)))  float    v8f;
typedef __attribute__((ext_vector_type(4)))  int      v4i;

#define B_    8
#define N_    4096
#define D_    768
#define H_    256
#define NROWS (B_ * N_)        // 32768 transform rows

#define NW      8              // waves per block (256 threads)
#define RPW     4              // rows per wave
#define THREADS (NW * 32)
#define NBLOCKS (NROWS / (NW * RPW))   // 1024

#define PI_F 3.14159265358979323846f

#if __has_builtin(__builtin_amdgcn_global_load_async_to_lds_b128)
#define HAVE_ASYNC 1
#else
#define HAVE_ASYNC 0
#endif

#if __has_builtin(__builtin_amdgcn_s_wait_asynccnt)
#define WAIT_ASYNC(n) __builtin_amdgcn_s_wait_asynccnt(n)
#else
#define WAIT_ASYNC(n) asm volatile("s_wait_asynccnt " #n)
#endif

typedef __attribute__((address_space(1))) v4i* gv4i_p;   // global int4*
typedef __attribute__((address_space(3))) v4i* lv4i_p;   // LDS int4*

union uh16 { v16h v; _Float16 e[16]; };
union uf8  { v8f  v; float    e[8];  };

__device__ __forceinline__ v8f wmma16(v16h a, v16h b, v8f c) {
  // (neg_a, A, neg_b, B, c_mod, C, reuse_a, reuse_b)
  return __builtin_amdgcn_wmma_f32_16x16x32_f16(false, a, false, b, (short)0, c,
                                                false, false);
}

// A-fragment half j -> K index within 0..31 (CDNA5 ISA 7.12.2, 16-bit A 16x32).
// Symmetric mapping assumed for B fragments (we control both sides of every
// constant fragment, so layouts stay self-consistent).
__device__ __forceinline__ int kfrag(int lane, int j) {
  return (j < 8 ? j : j + 8) + ((lane & 16) ? 8 : 0);
}

// Fragment stored linearly per lane: [lane][16] halves, 32B aligned.
__device__ __forceinline__ v16h ld_frag_lin(const _Float16* p) {
  return *(const v16h*)p;
}

// A fragment from a row-major [16][64] f16 LDS tile (K padded to 64).
__device__ __forceinline__ v16h ld_afrag(const _Float16* base, int lane, int kt) {
  const int M   = lane & 15;
  const int hi8 = (lane & 16) ? 8 : 0;
  const _Float16* p = base + M * 64 + kt * 32 + hi8;   // 16B aligned
  v8h lo = *(const v8h*)(p);
  v8h hh = *(const v8h*)(p + 16);
  return __builtin_shufflevector(lo, hh, 0, 1, 2, 3, 4, 5, 6, 7,
                                         8, 9, 10, 11, 12, 13, 14, 15);
}

#if HAVE_ASYNC
// Stream one 768-f32 row into LDS (6 x b128 per lane), tracked on ASYNCcnt.
__device__ __forceinline__ void issue_row_async(const float* g, float* lds, int l) {
#pragma unroll
  for (int c = 0; c < 6; ++c) {
    int off = (c * 32 + l) * 4;
    __builtin_amdgcn_global_load_async_to_lds_b128(
        (gv4i_p)(const_cast<float*>(g) + off),
        (lv4i_p)(lds + off),
        0, 0);
  }
}
#endif

// ---------------------------------------------------------------------------
// Context mean: c[b][d] = mean_n x[b][n][d]   (two-pass, deterministic)
// ---------------------------------------------------------------------------
__global__ void ctx_partial_kernel(const float* __restrict__ x,
                                   float* __restrict__ part) {
  const int b   = blockIdx.x;
  const int d   = blockIdx.y * 256 + threadIdx.x;
  const int seg = blockIdx.z;
  float s = 0.f;
#pragma unroll 4
  for (int i = 0; i < 256; ++i) {
    int n = seg * 256 + i;
    s += x[((size_t)b * N_ + n) * D_ + d];
  }
  part[(size_t)(b * 16 + seg) * D_ + d] = s;
}

__global__ void ctx_reduce_kernel(const float* __restrict__ part,
                                  float* __restrict__ cmean) {
  const int b = blockIdx.x;
  const int d = blockIdx.y * 256 + threadIdx.x;
  float s = 0.f;
#pragma unroll
  for (int g = 0; g < 16; ++g) s += part[(size_t)(b * 16 + g) * D_ + d];
  cmean[b * D_ + d] = s * (1.f / (float)N_);
}

// ---------------------------------------------------------------------------
// Tiny context MLP: delta[b][d] = gelu(c @ w1 + b1) @ w2 + b2
// ---------------------------------------------------------------------------
__global__ void mlp_kernel(const float* __restrict__ c,
                           const float* __restrict__ w1,
                           const float* __restrict__ b1,
                           const float* __restrict__ w2,
                           const float* __restrict__ b2,
                           float* __restrict__ delta) {
  __shared__ float sh[B_ * H_];
  const int t = threadIdx.x;
  for (int b = 0; b < B_; ++b) {
    float acc = b1[t];
    for (int k = 0; k < D_; ++k) acc += c[b * D_ + k] * w1[k * H_ + t];
    float xx = acc;
    float u = 0.7978845608028654f * (xx + 0.044715f * xx * xx * xx);
    sh[b * H_ + t] = 0.5f * xx * (1.f + tanhf(u));
  }
  __syncthreads();
  for (int j = 0; j < 3; ++j) {
    int d = j * 256 + t;
    for (int b = 0; b < B_; ++b) {
      float acc = b2[d];
      for (int k = 0; k < H_; ++k) acc += sh[b * H_ + k] * w2[k * D_ + d];
      delta[b * D_ + d] = acc;
    }
  }
}

// ---------------------------------------------------------------------------
// Fused FFT -> filter -> modReLU -> iFFT, one row per wave iteration.
// ---------------------------------------------------------------------------
__global__ void __launch_bounds__(THREADS)
fft_mixer_kernel(const float* __restrict__ x,
                 const float* __restrict__ Wb,
                 const float* __restrict__ bias,
                 const float* __restrict__ delta,
                 float* __restrict__ out) {
  // Constant fragments (B layout): f = part*6 + kt*3 + nt
  __shared__ __align__(32) _Float16 sBF[12][32][16];   // fwd DFT48 (K pad 64)
  __shared__ __align__(32) _Float16 sBI[12][32][16];   // inv DFT48 * 1/768
  __shared__ __align__(32) _Float16 sA16c[3][32][16];  // DFT16: Ar, Ai, -Ai
  __shared__ float2 sTw[768];                          // W768^{n1*k2}
  __shared__ __align__(16) _Float16 sRow[NW][2048];    // per-wave [2][16][64]
#if HAVE_ASYNC
  __shared__ __align__(16) float sStage[NW][2][D_];    // async row staging
#endif

  const int tid = threadIdx.x;
  const float scale = 1.f / (float)D_;

  // ---- build constant tables (per block) ----
  for (int i = tid; i < 12 * 32 * 16; i += THREADS) {
    int j = i & 15, lane = (i >> 4) & 31, f = i >> 9;   // f = part*6+kt*3+nt
    int nt = f % 3, kt = (f / 3) & 1, part = f / 6;
    int K = kt * 32 + kfrag(lane, j);
    int Nc = nt * 16 + (lane & 15);
    float ang = -2.f * PI_F * (float)(K * Nc) / 48.f;   // forward DFT48
    float s, cc; __sincosf(ang, &s, &cc);
    sBF[f][lane][j] = (_Float16)((K < 48) ? (part ? s : cc) : 0.f);
    float s2, c2; __sincosf(-ang, &s2, &c2);            // inverse, * 1/768
    sBI[f][lane][j] = (_Float16)((K < 48) ? (part ? (-s2 * scale)
                                                  : (c2 * scale)) : 0.f);
  }
  for (int i = tid; i < 3 * 32 * 16; i += THREADS) {
    int j = i & 15, lane = (i >> 4) & 31, which = i >> 9;
    int K = kfrag(lane, j);
    int M = lane & 15;
    float ang = 2.f * PI_F * (float)(M * K) / 16.f;     // DFT16 = e^{-i ang}
    float s, cc; __sincosf(ang, &s, &cc);
    float v = 0.f;
    if (K < 16) v = (which == 0) ? cc : ((which == 1) ? -s : s);
    sA16c[which][lane][j] = (_Float16)v;
  }
  for (int i = tid; i < 768; i += THREADS) {
    int n1 = i / 48, k2 = i % 48;
    float ang = -2.f * PI_F * (float)(n1 * k2) / (float)D_;
    float s, cc; __sincosf(ang, &s, &cc);
    sTw[i] = make_float2(cc, s);
  }
  __syncthreads();

  const int w = tid >> 5;
  const int l = tid & 31;
  const int l16 = l & 15;
  const int mb  = (l & 16) ? 8 : 0;
  _Float16* srow = &sRow[w][0];    // [0..1023] real part, [1024..2047] imag
  const int gw = blockIdx.x * NW + w;

  // One-time zero of the K=48..63 padding (never touched by per-row writes).
#pragma unroll
  for (int t = 0; t < 16; ++t) {
    int e = t * 32 + l;                    // 512 pad halves, both parts
    int part = e >> 8, rr = (e >> 4) & 15, cc = e & 15;
    srow[part * 1024 + rr * 64 + 48 + cc] = (_Float16)0.f;
  }

  // Loop-invariant DFT16 A-fragments.
  const v16h a16r  = ld_frag_lin(&sA16c[0][l][0]);
  const v16h a16i  = ld_frag_lin(&sA16c[1][l][0]);
  const v16h a16mi = ld_frag_lin(&sA16c[2][l][0]);

#if HAVE_ASYNC
  issue_row_async(x + (size_t)(gw * RPW) * D_, &sStage[w][0][0], l);
#endif

#pragma unroll 1
  for (int it = 0; it < RPW; ++it) {
    const int row = gw * RPW + it;
    const int b = row >> 12;
    const int n = row & 4095;
    const float* xr = x + (size_t)row * D_;
    const float* wrow = Wb + (size_t)n * D_;
    const float* drow = delta + b * D_;
    __builtin_prefetch(wrow, 0, 1);                    // warm filter row

#if HAVE_ASYNC
    if (it + 1 < RPW) {
      issue_row_async(xr + D_, &sStage[w][(it + 1) & 1][0], l);
      WAIT_ASYNC(6);                                   // current row resident
    } else {
      WAIT_ASYNC(0);
    }
    const float* src = &sStage[w][it & 1][0];
#else
    if (it + 1 < RPW) __builtin_prefetch(xr + D_, 0, 1);
    const float* src = xr;
#endif

    // ---- stage row as f16 [n1][n2], n2 padded to 64 ----
#pragma unroll
    for (int t = 0; t < 6; ++t) {
      float4 v4 = ((const float4*)src)[t * 32 + l];
      int nb = (t * 32 + l) * 4;
      srow[((nb + 0) & 15) * 64 + ((nb + 0) >> 4)] = (_Float16)v4.x;
      srow[((nb + 1) & 15) * 64 + ((nb + 1) >> 4)] = (_Float16)v4.y;
      srow[((nb + 2) & 15) * 64 + ((nb + 2) >> 4)] = (_Float16)v4.z;
      srow[((nb + 3) & 15) * 64 + ((nb + 3) >> 4)] = (_Float16)v4.w;
    }

    v16h xa0 = ld_afrag(srow, l, 0);
    v16h xa1 = ld_afrag(srow, l, 1);
    float* orow = out + (size_t)row * D_;

    // ---- fused per-k2-tile pipeline ----
#pragma unroll
    for (int nt = 0; nt < 3; ++nt) {
      const int k2 = nt * 16 + l16;

      // Stage A: Y = X @ DFT48 (real input)
      v8f yr = {};
      yr = wmma16(xa0, ld_frag_lin(&sBF[0 + nt][l][0]), yr);
      yr = wmma16(xa1, ld_frag_lin(&sBF[3 + nt][l][0]), yr);
      v8f yi = {};
      yi = wmma16(xa0, ld_frag_lin(&sBF[6 + nt][l][0]), yi);
      yi = wmma16(xa1, ld_frag_lin(&sBF[9 + nt][l][0]), yi);
      uf8 YR, YI; YR.v = yr; YI.v = yi;

      // Twiddle (keep tw in regs for conjugate reuse) -> Z B-frags
      uf8 TX, TY;
      uh16 Zr, Zi;
#pragma unroll
      for (int r = 0; r < 8; ++r) {
        float2 t = sTw[(r + mb) * 48 + k2];
        TX.e[r] = t.x; TY.e[r] = t.y;
        float a = YR.e[r], bb = YI.e[r];
        Zr.e[r] = (_Float16)(a * t.x - bb * t.y);
        Zi.e[r] = (_Float16)(a * t.y + bb * t.x);
        Zr.e[r + 8] = (_Float16)0.f;
        Zi.e[r + 8] = (_Float16)0.f;
      }

      // Stage C: F = DFT16 @ Z
      v8f fr = {};
      fr = wmma16(a16mi, Zi.v, fr);
      fr = wmma16(a16r,  Zr.v, fr);        // Fr = Ar.Zr - Ai.Zi
      v8f fi = {};
      fi = wmma16(a16i,  Zr.v, fi);
      fi = wmma16(a16r,  Zi.v, fi);        // Fi = Ar.Zi + Ai.Zr
      uf8 FR, FI; FR.v = fr; FI.v = fi;

      // Adaptive filter + modReLU (rsqrt form, no divide)
      uh16 FrB, FiB;
#pragma unroll
      for (int r = 0; r < 8; ++r) {
        int k = k2 + 48 * (r + mb);                    // frequency bin
        float wv = wrow[k] + drow[k];
        float pr = FR.e[r] * wv, pi = FI.e[r] * wv;
        float m2 = fmaxf(pr * pr + pi * pi, 1e-16f);
        float rs = __builtin_amdgcn_rsqf(m2);          // v_rsq_f32
        float mag = m2 * rs;                           // sqrt(m2) >= 1e-8
        float sc = fmaxf(mag + bias[k], 0.f) * rs;     // relu(mag+b)/mag
        FrB.e[r] = (_Float16)(pr * sc);
        FiB.e[r] = (_Float16)(pi * sc);
        FrB.e[r + 8] = (_Float16)0.f;
        FiB.e[r + 8] = (_Float16)0.f;
      }

      // Inverse Stage C': G = IDFT16 @ F  (IDFT16 = Ar - i*Ai, unscaled)
      v8f gr = {};
      gr = wmma16(a16i,  FiB.v, gr);
      gr = wmma16(a16r,  FrB.v, gr);       // Gr = Ar.Fr + Ai.Fi
      v8f gi = {};
      gi = wmma16(a16mi, FrB.v, gi);
      gi = wmma16(a16r,  FiB.v, gi);       // Gi = Ar.Fi - Ai.Fr
      uf8 GR, GI; GR.v = gr; GI.v = gi;

      // Conjugate twiddle (reuse TX/TY), transpose-store H into LDS
#pragma unroll
      for (int r = 0; r < 8; ++r) {
        int M = r + mb;
        float hr = GR.e[r] * TX.e[r] + GI.e[r] * TY.e[r];
        float hi = GI.e[r] * TX.e[r] - GR.e[r] * TY.e[r];
        srow[M * 64 + k2]        = (_Float16)hr;
        srow[1024 + M * 64 + k2] = (_Float16)hi;
      }
    }

    // ---- inverse Stage A': Y = Re( H @ IDFT48 ) ----
    v16h hr0 = ld_afrag(srow, l, 0);
    v16h hr1 = ld_afrag(srow, l, 1);
    v16h hi0 = ld_afrag(srow + 1024, l, 0);
    v16h hi1 = ld_afrag(srow + 1024, l, 1);
#pragma unroll
    for (int nt = 0; nt < 3; ++nt) {
      v8f y = {};
      y = wmma16(hr0, ld_frag_lin(&sBI[0 + nt][l][0]), y);
      y = wmma16(hr1, ld_frag_lin(&sBI[3 + nt][l][0]), y);
      y = wmma16(hi0, ld_frag_lin(&sBI[6 + nt][l][0]), y);
      y = wmma16(hi1, ld_frag_lin(&sBI[9 + nt][l][0]), y);
      uf8 Y; Y.v = y;
      int n2 = nt * 16 + l16;
      float4 lo4 = make_float4(Y.e[0], Y.e[1], Y.e[2], Y.e[3]);
      float4 hi4 = make_float4(Y.e[4], Y.e[5], Y.e[6], Y.e[7]);
      *(float4*)(orow + 16 * n2 + mb)     = lo4;   // y[n1 + 16*n2], n1=M
      *(float4*)(orow + 16 * n2 + mb + 4) = hi4;
    }
  }
}

// ---------------------------------------------------------------------------
extern "C" void kernel_launch(void* const* d_in, const int* in_sizes, int n_in,
                              void* d_out, int out_size, void* d_ws, size_t ws_size,
                              hipStream_t stream) {
  (void)in_sizes; (void)n_in; (void)out_size; (void)ws_size;
  const float* x    = (const float*)d_in[0];   // (8,4096,768)
  const float* Wb   = (const float*)d_in[1];   // (4096,768)
  const float* bias = (const float*)d_in[2];   // (768,)
  const float* w1   = (const float*)d_in[3];   // (768,256)
  const float* b1   = (const float*)d_in[4];   // (256,)
  const float* w2   = (const float*)d_in[5];   // (256,768)
  const float* b2   = (const float*)d_in[6];   // (768,)
  float* out = (float*)d_out;

  float* ws    = (float*)d_ws;
  float* part  = ws;                       // 16*8*768
  float* cmean = part + 16 * B_ * D_;      // 8*768
  float* delta = cmean + B_ * D_;          // 8*768

  ctx_partial_kernel<<<dim3(B_, 3, 16), 256, 0, stream>>>(x, part);
  ctx_reduce_kernel <<<dim3(B_, 3),     256, 0, stream>>>(part, cmean);
  mlp_kernel        <<<1, 256, 0, stream>>>(cmean, w1, b1, w2, b2, delta);
  fft_mixer_kernel  <<<NBLOCKS, THREADS, 0, stream>>>(x, Wb, bias, delta, out);
}
